// MultiHeadedAttention_41824391528449
// MI455X (gfx1250) — matrix operand order repaired and checked by reading, not compiled, not run on hardware
//
#include <hip/hip_runtime.h>
#include <hip/hip_bf16.h>

typedef unsigned short u16s;
typedef __attribute__((ext_vector_type(16))) __bf16          v16bf;
typedef __attribute__((ext_vector_type(8)))  float           v8f;
typedef __attribute__((ext_vector_type(8)))  unsigned short  u16x8;
typedef __attribute__((ext_vector_type(4)))  unsigned int    u32x4;
typedef __attribute__((ext_vector_type(8)))  int             i32x8;
typedef __attribute__((ext_vector_type(4)))  int             i32x4;

#define BATCH 4
#define SEQ   2048
#define EMB   1024
#define NHEAD 16
#define HDIM  64
#define SW    (SEQ / 32)   // mask words per row

union Frag2 { v16bf v; u16x8 h[2]; u16s s[16]; };

__device__ inline u16s f2bf(float f) {
    union { __bf16 h; u16s s; } u;
    u.h = (__bf16)f;
    return u.s;
}

__device__ inline v8f wmma_bf16(v16bf a, v16bf b, v8f c) {
    return __builtin_amdgcn_wmma_f32_16x16x32_bf16(false, a, false, b, (short)0, c, false, false);
}

// all-ones bf16 fragment (rowsum = P x 1 via WMMA)
__device__ inline v16bf ones_frag() {
    Frag2 u;
#pragma unroll
    for (int i = 0; i < 16; ++i) u.s[i] = 0x3F80;
    return u.v;
}

// in-row (16-lane) xor butterfly via v_permlane16_b32
__device__ inline float permlane16_f(float v, unsigned s0, unsigned s1) {
    unsigned x = __float_as_uint(v);
    return __uint_as_float(__builtin_amdgcn_permlane16(x, x, s0, s1, false, false));
}
__device__ inline float rowmax16(float best) {
    best = fmaxf(best, permlane16_f(best, 0x67452301u, 0xEFCDAB89u));  // xor 1
    best = fmaxf(best, permlane16_f(best, 0x45670123u, 0xCDEF89ABu));  // xor 3
    best = fmaxf(best, permlane16_f(best, 0x01234567u, 0x89ABCDEFu));  // xor 7
    best = fmaxf(best, permlane16_f(best, 0xFEDCBA98u, 0x76543210u));  // xor 8
    return best;
}

// A-matrix 16x32 bf16 fragment: lane&15 = row M; per lane two contiguous 16B K-runs.
__device__ inline v16bf load_a_frag(const u16s* base, int row_stride, int lane, int kbase) {
    Frag2 u;
    const u16s* rp = base + (size_t)(lane & 15) * row_stride + ((lane < 16) ? 0 : 8) + kbase;
    u.h[0] = *(const u16x8*)(rp);
    u.h[1] = *(const u16x8*)(rp + 16);
    return u.v;
}
// B-matrix 32x16 bf16 fragment from [N][K] storage: lane&15 = col N; per lane two contiguous 16B K-runs.
__device__ inline v16bf load_b_frag_t(const u16s* base, int n_stride, int lane, int kbase) {
    Frag2 u;
    const u16s* np = base + (size_t)(lane & 15) * n_stride + ((lane < 16) ? 0 : 8) + kbase;
    u.h[0] = *(const u16x8*)(np);
    u.h[1] = *(const u16x8*)(np + 16);
    return u.v;
}

// TDM: 2-D bf16 tile load Global->LDS (D# per CDNA5 ISA).
// tile_d0/tile_d1 in elements; stride0 = tensor dim0 stride in elements.
__device__ inline void tdm_load_tile_2d(const u16s* gptr, unsigned lds_addr,
                                        int tile_d0, int tile_d1,
                                        int tdim0, int tdim1, int stride0) {
    unsigned long long ga = (unsigned long long)(size_t)gptr;
    u32x4 g0;
    g0[0] = 1u;                                            // count=1, user descriptor
    g0[1] = lds_addr;                                      // LDS byte address
    g0[2] = (unsigned)(ga & 0xFFFFFFFFu);                  // global_addr[31:0]
    g0[3] = (unsigned)((ga >> 32) & 0x1FFFFFFu) | (2u << 30); // global_addr[56:32] | type=2
    i32x8 g1;
    g1[0] = (1 << 16);                                     // wg_mask=0 | data_size=1 (2B)
    g1[1] = (tdim0 & 0xFFFF) << 16;                        // ab_addr=0 | tensor_dim0[15:0]
    g1[2] = ((tdim0 >> 16) & 0xFFFF) | ((tdim1 & 0xFFFF) << 16);   // dim0 hi | dim1 lo
    g1[3] = ((tdim1 >> 16) & 0xFFFF) | ((tile_d0 & 0xFFFF) << 16); // dim1 hi | tile_dim0
    g1[4] = (tile_d1 & 0xFFFF);                            // tile_dim1 | tile_dim2=0
    g1[5] = stride0;                                       // tensor_dim0_stride[31:0]
    g1[6] = 0;                                             // stride0 hi | dim1_stride lo
    g1[7] = 0;
    i32x4 z4 = {0, 0, 0, 0};                               // groups 2/3 unused (2-D)
    i32x8 z8 = {0, 0, 0, 0, 0, 0, 0, 0};
    __builtin_amdgcn_tensor_load_to_lds(g0, g1, z4, z4, z8, 0);   // 6-arg form
}

// fp32 [K][N] -> bf16 [N][K]
__global__ __launch_bounds__(256) void convert_transpose_kernel(const float* __restrict__ src,
                                                                u16s* __restrict__ dst,
                                                                int K, int N) {
    int i = blockIdx.x * 256 + threadIdx.x;
    if (i < K * N) {
        int k = i / N, n = i % N;
        dst[(size_t)n * K + k] = f2bf(src[i]);
    }
}

// int32 mask -> bit mask via wave32 ballot
__global__ __launch_bounds__(256) void pack_mask_kernel(const int* __restrict__ mask,
                                                        unsigned* __restrict__ bits,
                                                        unsigned n_words) {
    size_t w = (size_t)blockIdx.x * 8 + (threadIdx.x >> 5);
    size_t i = w * 32 + (threadIdx.x & 31);
    unsigned long long bal = __ballot(w < n_words && mask[i] != 0);
    if ((threadIdx.x & 31) == 0 && w < n_words) bits[w] = (unsigned)bal;
}

// C[M,N] = (A[M,K] @ W[K,N] + bias) * oscale, W supplied transposed ([N][K] bf16).
// A staged by threads (fp32->bf16 convert); B tile DMA'd by the Tensor Data Mover.
template <bool IN_BF16, bool SPLIT_HEADS>
__global__ __launch_bounds__(256) void gemm_kernel(const void* __restrict__ Aptr,
                                                   const u16s* __restrict__ Wt,
                                                   const float* __restrict__ bias,
                                                   void* __restrict__ Cptr,
                                                   int M, int N, int K, float oscale) {
    __shared__ __align__(16) u16s As[2][128][64];
    __shared__ __align__(16) u16s Bt[2][128][64];

    const int tid  = threadIdx.x;
    const int lane = tid & 31;
    const int wave = tid >> 5;
    const int wm   = wave >> 1;
    const int wn   = wave & 1;
    const int m0   = blockIdx.y * 128;
    const int n0   = blockIdx.x * 128;

    const int srow = tid >> 1, scol = (tid & 1) * 32;

    auto stageA = [&](int buf, int k0) {
        if (IN_BF16) {
            const u16x8* Ah = (const u16x8*)((const u16s*)Aptr + (size_t)(m0 + srow) * K + k0 + scol);
#pragma unroll
            for (int j = 0; j < 4; ++j) *(u16x8*)&As[buf][srow][scol + 8 * j] = Ah[j];
            if (k0 + 128 < K) __builtin_prefetch((const u16s*)Ah + 128, 0, 1);
        } else {
            const float4* Af = (const float4*)((const float*)Aptr + (size_t)(m0 + srow) * K + k0 + scol);
            alignas(16) u16s tmp[32];
#pragma unroll
            for (int j = 0; j < 8; ++j) {
                float4 f = Af[j];
                tmp[4 * j + 0] = f2bf(f.x);
                tmp[4 * j + 1] = f2bf(f.y);
                tmp[4 * j + 2] = f2bf(f.z);
                tmp[4 * j + 3] = f2bf(f.w);
            }
#pragma unroll
            for (int j = 0; j < 4; ++j) *(u16x8*)&As[buf][srow][scol + 8 * j] = *(const u16x8*)&tmp[8 * j];
            if (k0 + 128 < K) __builtin_prefetch((const float*)Af + 128, 0, 1);
        }
    };
    auto stageB = [&](int buf, int k0) {
        if (wave == 0)   // one wave drives the TDM; EXEC-independent DMA
            tdm_load_tile_2d(Wt + (size_t)n0 * K + k0, (unsigned)(size_t)&Bt[buf][0][0],
                             64, 128, 64, 128, K);
    };

    v8f acc[2][4] = {};

    stageA(0, 0);
    stageB(0, 0);
    if (wave == 0) __builtin_amdgcn_s_wait_tensorcnt(0);
    __syncthreads();

    int buf = 0;
    for (int k0 = 0; k0 < K; k0 += 64) {
        if (k0 + 64 < K) { stageA(buf ^ 1, k0 + 64); stageB(buf ^ 1, k0 + 64); }

        v16bf a[2][2], b[2][4];
#pragma unroll
        for (int ks = 0; ks < 2; ++ks)
#pragma unroll
            for (int mt = 0; mt < 2; ++mt)
                a[ks][mt] = load_a_frag(&As[buf][wm * 32 + mt * 16][0], 64, lane, ks * 32);
#pragma unroll
        for (int ks = 0; ks < 2; ++ks)
#pragma unroll
            for (int nt = 0; nt < 4; ++nt)
                b[ks][nt] = load_b_frag_t(&Bt[buf][wn * 64 + nt * 16][0], 64, lane, ks * 32);
#pragma unroll
        for (int ks = 0; ks < 2; ++ks)
#pragma unroll
            for (int mt = 0; mt < 2; ++mt)
#pragma unroll
                for (int nt = 0; nt < 4; ++nt)
                    acc[mt][nt] = wmma_bf16(a[ks][mt], b[ks][nt], acc[mt][nt]);

        if (wave == 0) __builtin_amdgcn_s_wait_tensorcnt(0);
        __syncthreads();
        buf ^= 1;
    }

#pragma unroll
    for (int mt = 0; mt < 2; ++mt)
#pragma unroll
        for (int nt = 0; nt < 4; ++nt)
#pragma unroll
            for (int e = 0; e < 8; ++e) {
                int row = m0 + wm * 32 + mt * 16 + e + ((lane >= 16) ? 8 : 0);
                int col = n0 + wn * 64 + nt * 16 + (lane & 15);
                float v = (acc[mt][nt][e] + bias[col]) * oscale;
                if (SPLIT_HEADS) {
                    int b = row >> 11, s = row & (SEQ - 1);
                    int h = col >> 6, d = col & (HDIM - 1);
                    ((u16s*)Cptr)[(((size_t)b * NHEAD + h) * SEQ + s) * HDIM + d] = f2bf(v);
                } else {
                    ((float*)Cptr)[(size_t)row * N + col] = v;
                }
            }
}

// Flash attention: block = 128 query rows of one (b,h); each wave owns 16 rows.
// K tile via TDM; V tile transposed by threads; rowsum via WMMA; rowmax via permlane16.
__global__ __launch_bounds__(256) void attn_kernel(const u16s* __restrict__ Q,
                                                   const u16s* __restrict__ Kb_,
                                                   const u16s* __restrict__ Vb_,
                                                   const unsigned* __restrict__ mbits,
                                                   u16s* __restrict__ ctx) {
    __shared__ __align__(16) u16s Ks[2][64][64];   // [key][d]
    __shared__ __align__(16) u16s Vt[2][64][64];   // [d][key]
    __shared__ __align__(16) u16s Ps[8][16][64];   // per-wave P staging [row][key]

    const int tid  = threadIdx.x;
    const int lane = tid & 31;
    const int wave = tid >> 5;
    const int bh   = blockIdx.x;
    const int b    = bh >> 4;
    const int h    = bh & 15;
    const int q0   = blockIdx.y * 128 + wave * 16;

    const u16s* Qbase = Q   + ((size_t)bh * SEQ + q0) * HDIM;
    const u16s* Kbase = Kb_ + (size_t)bh * SEQ * HDIM;
    const u16s* Vbase = Vb_ + (size_t)bh * SEQ * HDIM;
    const unsigned* mbase = mbits + (size_t)b * SEQ * SW;

    v16bf qa[2];
    qa[0] = load_a_frag(Qbase, HDIM, lane, 0);
    qa[1] = load_a_frag(Qbase, HDIM, lane, 32);
    const v16bf onesf = ones_frag();

    v8f   oacc[4] = {};
    v8f   lacc    = {};
    float mrow[8];
#pragma unroll
    for (int e = 0; e < 8; ++e) mrow[e] = -3.0e38f;

    const int srow = tid >> 2, scol = (tid & 3) * 16;

    auto stage = [&](int buf, int kb0) {
        if (wave == 0)   // K tile: straight 2-D copy -> TDM
            tdm_load_tile_2d(Kbase + (size_t)kb0 * HDIM, (unsigned)(size_t)&Ks[buf][0][0],
                             HDIM, 64, HDIM, SEQ, HDIM);
        // V tile transposed by threads (TDM cannot transpose)
        const u16s* vg = Vbase + (size_t)(kb0 + srow) * HDIM + scol;
        u16x8 v0 = *(const u16x8*)vg;
        u16x8 v1 = *(const u16x8*)(vg + 8);
#pragma unroll
        for (int i = 0; i < 8; ++i) {
            Vt[buf][scol + i][srow]     = v0[i];
            Vt[buf][scol + 8 + i][srow] = v1[i];
        }
        if (kb0 + 64 < SEQ) __builtin_prefetch(vg + 64 * HDIM, 0, 1);
    };

    stage(0, 0);
    if (wave == 0) __builtin_amdgcn_s_wait_tensorcnt(0);
    __syncthreads();

    int buf = 0;
    for (int kb0 = 0; kb0 < SEQ; kb0 += 64) {
        if (kb0 + 64 < SEQ) stage(buf ^ 1, kb0 + 64);

        // ---- scores ----
        v16bf kfr[2][4];
#pragma unroll
        for (int f = 0; f < 2; ++f)
#pragma unroll
            for (int nt = 0; nt < 4; ++nt)
                kfr[f][nt] = load_b_frag_t(&Ks[buf][nt * 16][0], 64, lane, f * 32);
        v8f sc[4] = {};
#pragma unroll
        for (int f = 0; f < 2; ++f)
#pragma unroll
            for (int nt = 0; nt < 4; ++nt)
                sc[nt] = wmma_bf16(qa[f], kfr[f][nt], sc[nt]);

        // ---- V fragments issued early ----
        v16bf vfr[2][4];
#pragma unroll
        for (int f = 0; f < 2; ++f)
#pragma unroll
            for (int dt = 0; dt < 4; ++dt)
                vfr[f][dt] = load_b_frag_t(&Vt[buf][dt * 16][0], 64, lane, f * 32);

        // ---- bit-mask + row-max ----
        float rm[8];
#pragma unroll
        for (int e = 0; e < 8; ++e) {
            int qrow = q0 + e + ((lane >= 16) ? 8 : 0);
            uint2 w = *(const uint2*)(mbase + (size_t)qrow * SW + (kb0 >> 5));
            float best = -3.0e38f;
#pragma unroll
            for (int nt = 0; nt < 4; ++nt) {
                unsigned ww = (nt < 2) ? w.x : w.y;
                int bitpos  = (nt & 1) * 16 + (lane & 15);
                float s = sc[nt][e];
                s = ((ww >> bitpos) & 1u) ? s : -1.0e9f;
                sc[nt][e] = s;
                best = fmaxf(best, s);
            }
            rm[e] = rowmax16(best);
        }

        // ---- online softmax ----
#pragma unroll
        for (int e = 0; e < 8; ++e) {
            float mn    = fmaxf(mrow[e], rm[e]);
            float alpha = __expf(mrow[e] - mn);
            mrow[e]     = mn;
#pragma unroll
            for (int nt = 0; nt < 4; ++nt)
                sc[nt][e] = __expf(sc[nt][e] - mn);
#pragma unroll
            for (int dt = 0; dt < 4; ++dt) oacc[dt][e] *= alpha;
            lacc[e] *= alpha;
        }

        // ---- stage P ----
#pragma unroll
        for (int nt = 0; nt < 4; ++nt)
#pragma unroll
            for (int e = 0; e < 8; ++e) {
                int r = e + ((lane >= 16) ? 8 : 0);
                int c = nt * 16 + (lane & 15);
                Ps[wave][r][c] = f2bf(sc[nt][e]);
            }

        // ---- O += P x V ; rowsum += P x 1 ----
#pragma unroll
        for (int f = 0; f < 2; ++f) {
            v16bf pa = load_a_frag(&Ps[wave][0][0], 64, lane, f * 32);
            lacc = wmma_bf16(pa, onesf, lacc);
#pragma unroll
            for (int dt = 0; dt < 4; ++dt)
                oacc[dt] = wmma_bf16(pa, vfr[f][dt], oacc[dt]);
        }
        if (wave == 0) __builtin_amdgcn_s_wait_tensorcnt(0);
        __syncthreads();
        buf ^= 1;
    }

    // ---- finalize ----
#pragma unroll
    for (int e = 0; e < 8; ++e) {
        float inv  = 1.0f / lacc[e];
        int   qrow = q0 + e + ((lane >= 16) ? 8 : 0);
#pragma unroll
        for (int dt = 0; dt < 4; ++dt) {
            int d = dt * 16 + (lane & 15);
            ctx[((size_t)b * SEQ + qrow) * EMB + h * HDIM + d] = f2bf(oacc[dt][e] * inv);
        }
    }
}

extern "C" void kernel_launch(void* const* d_in, const int* in_sizes, int n_in,
                              void* d_out, int out_size, void* d_ws, size_t ws_size,
                              hipStream_t stream) {
    const float* query = (const float*)d_in[0];
    const float* key   = (const float*)d_in[1];
    const float* value = (const float*)d_in[2];
    const int*   mask  = (const int*)d_in[3];
    const float* Wq    = (const float*)d_in[4];
    const float* bq    = (const float*)d_in[5];
    const float* Wk    = (const float*)d_in[6];
    const float* bk    = (const float*)d_in[7];
    const float* Wv    = (const float*)d_in[8];
    const float* bv    = (const float*)d_in[9];
    const float* Wo    = (const float*)d_in[10];
    const float* bo    = (const float*)d_in[11];

    const size_t WEL  = (size_t)EMB * EMB;
    const size_t QKVE = (size_t)BATCH * NHEAD * SEQ * HDIM;
    const size_t CTXE = (size_t)BATCH * SEQ * EMB;

    u16s* ws   = (u16s*)d_ws;
    u16s* wqb  = ws;
    u16s* wkb  = ws + WEL;
    u16s* wvb  = ws + 2 * WEL;
    u16s* wob  = ws + 3 * WEL;
    u16s* qbuf = ws + 4 * WEL;
    u16s* kbuf = qbuf + QKVE;
    u16s* vbuf = kbuf + QKVE;
    u16s* ctx  = vbuf + QKVE;
    unsigned* mbits = (unsigned*)(ctx + CTXE);

    const int M = BATCH * SEQ, N = EMB, K = EMB;

    convert_transpose_kernel<<<(int)(WEL / 256), 256, 0, stream>>>(Wq, wqb, K, N);
    convert_transpose_kernel<<<(int)(WEL / 256), 256, 0, stream>>>(Wk, wkb, K, N);
    convert_transpose_kernel<<<(int)(WEL / 256), 256, 0, stream>>>(Wv, wvb, K, N);
    convert_transpose_kernel<<<(int)(WEL / 256), 256, 0, stream>>>(Wo, wob, K, N);
    {
        unsigned n_words = (unsigned)((size_t)BATCH * SEQ * SEQ / 32);
        pack_mask_kernel<<<(int)(n_words / 8), 256, 0, stream>>>(mask, mbits, n_words);
    }

    dim3 gg(N / 128, M / 128);
    gemm_kernel<false, true><<<gg, 256, 0, stream>>>(query, wqb, bq, qbuf, M, N, K, 0.125f);
    gemm_kernel<false, true><<<gg, 256, 0, stream>>>(key,   wkb, bk, kbuf, M, N, K, 1.0f);
    gemm_kernel<false, true><<<gg, 256, 0, stream>>>(value, wvb, bv, vbuf, M, N, K, 1.0f);

    attn_kernel<<<dim3(BATCH * NHEAD, SEQ / 128), 256, 0, stream>>>(qbuf, kbuf, vbuf, mbits, ctx);

    gemm_kernel<true, false><<<gg, 256, 0, stream>>>(ctx, wob, bo, d_out, M, N, K, 1.0f);
}
